// BERT_CRF_31903017074730
// MI455X (gfx1250) — matrix operand order repaired and checked by reading, not compiled, not run on hardware
//
#include <hip/hip_runtime.h>

// ---------------- problem constants ----------------
#define BB 8
#define SS 512
#define HH 768
#define NLAYER 12
#define NHEAD 12
#define DH 64
#define FFN 3072
#define LL 9
#define MTOK (BB * SS)   // 4096 tokens

// ---------------- types ----------------
typedef __bf16 bf16_t;
typedef __attribute__((ext_vector_type(16))) bf16_t bf16x16;
typedef __attribute__((ext_vector_type(16))) unsigned short ushortx16;
typedef __attribute__((ext_vector_type(8))) float floatx8;

union Frag { ushortx16 u; bf16x16 b; };
static_assert(sizeof(Frag) == 32, "frag size");

// native f32 -> bf16 (RNE in hardware) via __bf16 cast
__device__ __forceinline__ unsigned short f2bf(float f) {
  union { bf16_t h; unsigned short u; } cv;
  cv.h = (bf16_t)f;
  return cv.u;
}
__device__ __forceinline__ float bf2f(unsigned short h) {
  return __uint_as_float(((unsigned int)h) << 16);
}

// per-lane async copy 16B global -> LDS (CDNA5 ASYNCcnt path)
__device__ __forceinline__ void async_g2l_b128(const void* gaddr, void* lds) {
  unsigned int loff = (unsigned int)(unsigned long long)lds;  // low 32b = LDS offset
  asm volatile("global_load_async_to_lds_b128 %0, %1, off"
               :: "v"(loff), "v"((unsigned long long)gaddr)
               : "memory");
}
__device__ __forceinline__ void wait_async0() {
  asm volatile("s_wait_asynccnt 0x0" ::: "memory");
}

// ---------------- fp32 -> bf16 cast ----------------
__global__ void cast_f32_to_bf16_k(const float* __restrict__ in,
                                   unsigned short* __restrict__ out, size_t n) {
  size_t i = (size_t)blockIdx.x * blockDim.x + threadIdx.x;
  size_t stride = (size_t)gridDim.x * blockDim.x;
  for (; i < n; i += stride) out[i] = f2bf(in[i]);
}

// ---------------- tiled bf16 WMMA GEMM: C[M,N] = A[M,K] @ B[K,N] + bias ----------------
// EPI: 0 = fp32 out, 1 = bf16 out, 2 = gelu(x) -> bf16 out
// Block = 128 threads (4 waves), tile 64x64, BK = 32 (= WMMA K).
// Global->LDS staging uses global_load_async_to_lds_b128 (ASYNCcnt).
template <int EPI>
__global__ __launch_bounds__(128) void gemm_bf16_wmma(
    const unsigned short* __restrict__ A, const unsigned short* __restrict__ Bw,
    const float* __restrict__ bias, void* __restrict__ Cout, int M, int N, int K) {
  __shared__ unsigned short As[64][40];  // row stride 80B (16B aligned)
  __shared__ unsigned short Bs[32][72];  // row stride 144B (16B aligned)

  const int tid = threadIdx.x;
  const int lane = tid & 31;
  const int wv = tid >> 5;
  const int hl = lane >> 4;   // lane half (K split)
  const int lr = lane & 15;   // row/col within frag
  const int m0 = blockIdx.y * 64;
  const int n0 = blockIdx.x * 64;
  const int wm = (wv >> 1) * 32;
  const int wn = (wv & 1) * 32;

  floatx8 acc[2][2];
#pragma unroll
  for (int i = 0; i < 2; ++i)
#pragma unroll
    for (int j = 0; j < 2; ++j) acc[i][j] = (floatx8)0.0f;

  const bool bvec = ((N & 7) == 0) && (n0 + 64 <= N);

  for (int k0 = 0; k0 < K; k0 += 32) {
    // ---- A tile 64x32 (always in-bounds: M%64==0, K%32==0): async 16B/lane x2
#pragma unroll
    for (int p = 0; p < 2; ++p) {
      int r = p * 32 + (tid >> 2);       // 4 lanes per 64B row
      int cc = (tid & 3) * 8;
      const unsigned short* g = A + (size_t)(m0 + r) * K + k0 + cc;
      async_g2l_b128(g, &As[r][cc]);
    }
    // ---- B tile 32x64
    if (bvec) {
#pragma unroll
      for (int p = 0; p < 2; ++p) {
        int r = p * 16 + (tid >> 3);     // 8 lanes per 128B row
        int cc = (tid & 7) * 8;
        const unsigned short* g = Bw + (size_t)(k0 + r) * N + n0 + cc;
        async_g2l_b128(g, &Bs[r][cc]);
      }
    } else {  // edge path (classifier N=9), zero-filled scalar stores
#pragma unroll
      for (int p = 0; p < 16; ++p) {
        int idx = p * 128 + tid;
        int r = idx >> 6, cc = idx & 63;
        unsigned short v = 0;
        if (n0 + cc < N) v = Bw[(size_t)(k0 + r) * N + n0 + cc];
        Bs[r][cc] = v;
      }
    }
    wait_async0();
    __syncthreads();

    if (k0 + 32 < K)  // lowers to global_prefetch_b8
      __builtin_prefetch(A + (size_t)(m0 + (tid >> 3)) * K + k0 + 32, 0, 1);

    // ---- fragments (ISA 16-bit A 16x32 / B 32x16 lane layouts)
    Frag af[2], bfm[2];
#pragma unroll
    for (int i = 0; i < 2; ++i) {
      int row = wm + i * 16 + lr;
#pragma unroll
      for (int j = 0; j < 8; ++j) {
        int kk = (j < 4) ? (hl * 8 + 2 * j) : (16 + hl * 8 + 2 * (j - 4));
        unsigned int pv = *(const unsigned int*)&As[row][kk];
        af[i].u[2 * j] = (unsigned short)(pv & 0xFFFFu);
        af[i].u[2 * j + 1] = (unsigned short)(pv >> 16);
      }
    }
#pragma unroll
    for (int j2 = 0; j2 < 2; ++j2) {
      int col = wn + j2 * 16 + lr;
#pragma unroll
      for (int j = 0; j < 8; ++j) {
        int kk = (j < 4) ? (hl * 8 + 2 * j) : (16 + hl * 8 + 2 * (j - 4));
        bfm[j2].u[2 * j] = Bs[kk][col];
        bfm[j2].u[2 * j + 1] = Bs[kk + 1][col];
      }
    }
#pragma unroll
    for (int i = 0; i < 2; ++i)
#pragma unroll
      for (int j2 = 0; j2 < 2; ++j2)
        acc[i][j2] = __builtin_amdgcn_wmma_f32_16x16x32_bf16(
            false, af[i].b, false, bfm[j2].b, (short)0, acc[i][j2], false, false);
    __syncthreads();
  }

  // ---- epilogue (C layout: lane col = lr, VGPR r -> row r + 8*hl)
#pragma unroll
  for (int j2 = 0; j2 < 2; ++j2) {
    int n = n0 + wn + j2 * 16 + lr;
    if (n >= N) continue;
    float bv = bias ? bias[n] : 0.0f;
#pragma unroll
    for (int i = 0; i < 2; ++i) {
#pragma unroll
      for (int r = 0; r < 8; ++r) {
        int m = m0 + wm + i * 16 + hl * 8 + r;
        float v = acc[i][j2][r] + bv;
        if (EPI == 0) {
          ((float*)Cout)[(size_t)m * N + n] = v;
        } else if (EPI == 1) {
          ((unsigned short*)Cout)[(size_t)m * N + n] = f2bf(v);
        } else {
          float g = 0.5f * v * (1.0f + erff(v * 0.70710678118f));
          ((unsigned short*)Cout)[(size_t)m * N + n] = f2bf(g);
        }
      }
    }
  }
}

// ---------------- attention scores: S[b,h,i,j] = (q . k)/8 + mask-bias, bf16 out ----------------
__global__ __launch_bounds__(32) void attn_scores_k(
    const unsigned short* __restrict__ q, const unsigned short* __restrict__ k,
    const int* __restrict__ amask, unsigned short* __restrict__ attn) {
  const int lane = threadIdx.x, hl = lane >> 4, lr = lane & 15;
  const int ti = blockIdx.x, tj = blockIdx.y, bh = blockIdx.z;
  const int b = bh / NHEAD, h = bh % NHEAD;
  const int i = ti * 16 + lr;
  const int j = tj * 16 + lr;
  const unsigned short* qrow = q + (size_t)(b * SS + i) * HH + h * DH;
  const unsigned short* krow = k + (size_t)(b * SS + j) * HH + h * DH;
  floatx8 acc = (floatx8)0.0f;
#pragma unroll
  for (int kb = 0; kb < DH; kb += 32) {
    Frag a, bb;
#pragma unroll
    for (int t = 0; t < 8; ++t) {
      int kk = kb + ((t < 4) ? (hl * 8 + 2 * t) : (16 + hl * 8 + 2 * (t - 4)));
      unsigned int pv = *(const unsigned int*)&qrow[kk];
      a.u[2 * t] = (unsigned short)(pv & 0xFFFFu);
      a.u[2 * t + 1] = (unsigned short)(pv >> 16);
    }
#pragma unroll
    for (int t = 0; t < 8; ++t) {
      int kk = kb + ((t < 4) ? (hl * 8 + 2 * t) : (16 + hl * 8 + 2 * (t - 4)));
      unsigned int pv = *(const unsigned int*)&krow[kk];
      bb.u[2 * t] = (unsigned short)(pv & 0xFFFFu);
      bb.u[2 * t + 1] = (unsigned short)(pv >> 16);
    }
    acc = __builtin_amdgcn_wmma_f32_16x16x32_bf16(false, a.b, false, bb.b,
                                                  (short)0, acc, false, false);
  }
  const int col = tj * 16 + lr;
  const float biasv = (1.0f - (float)amask[b * SS + col]) * -10000.0f;
  unsigned short* orow = attn + (size_t)bh * SS * SS;
#pragma unroll
  for (int r = 0; r < 8; ++r) {
    int row = ti * 16 + hl * 8 + r;
    orow[(size_t)row * SS + col] = f2bf(acc[r] * 0.125f + biasv);
  }
}

// ---------------- softmax over last dim, in place on bf16 rows (1 wave / row) -------------
__global__ __launch_bounds__(32) void softmax_rows_k(unsigned short* __restrict__ attn) {
  unsigned short* p = attn + (size_t)blockIdx.x * SS;
  const int lane = threadIdx.x;
  float vals[SS / 32];
  float mx = -3.0e38f;
#pragma unroll
  for (int t = 0; t < SS / 32; ++t) {
    float v = bf2f(p[lane + t * 32]);
    vals[t] = v;
    mx = fmaxf(mx, v);
  }
#pragma unroll
  for (int o = 16; o > 0; o >>= 1) mx = fmaxf(mx, __shfl_xor(mx, o, 32));
  float s = 0.0f;
#pragma unroll
  for (int t = 0; t < SS / 32; ++t) {
    float e = __expf(vals[t] - mx);
    vals[t] = e;
    s += e;
  }
#pragma unroll
  for (int o = 16; o > 0; o >>= 1) s += __shfl_xor(s, o, 32);
  float inv = 1.0f / s;
#pragma unroll
  for (int t = 0; t < SS / 32; ++t) p[lane + t * 32] = f2bf(vals[t] * inv);
}

// ---------------- attention context: ctx[b,i,h,d] = sum_j attn * v ----------------
__global__ __launch_bounds__(32) void attn_ctx_k(
    const unsigned short* __restrict__ attn, const unsigned short* __restrict__ v,
    unsigned short* __restrict__ ctx) {
  const int lane = threadIdx.x, hl = lane >> 4, lr = lane & 15;
  const int ti = blockIdx.x, td = blockIdx.y, bh = blockIdx.z;
  const int b = bh / NHEAD, h = bh % NHEAD;
  const int i = ti * 16 + lr;
  const int d = td * 16 + lr;
  const unsigned short* arow = attn + ((size_t)bh * SS + i) * SS;
  floatx8 acc = (floatx8)0.0f;
  for (int k0 = 0; k0 < SS; k0 += 32) {
    Frag a, bb;
#pragma unroll
    for (int t = 0; t < 8; ++t) {
      int kk = k0 + ((t < 4) ? (hl * 8 + 2 * t) : (16 + hl * 8 + 2 * (t - 4)));
      unsigned int pv = *(const unsigned int*)&arow[kk];
      a.u[2 * t] = (unsigned short)(pv & 0xFFFFu);
      a.u[2 * t + 1] = (unsigned short)(pv >> 16);
    }
#pragma unroll
    for (int t = 0; t < 8; ++t) {
      int kk = k0 + ((t < 4) ? (hl * 8 + 2 * t) : (16 + hl * 8 + 2 * (t - 4)));
      bb.u[2 * t] = v[(size_t)(b * SS + kk) * HH + h * DH + d];
      bb.u[2 * t + 1] = v[(size_t)(b * SS + kk + 1) * HH + h * DH + d];
    }
    acc = __builtin_amdgcn_wmma_f32_16x16x32_bf16(false, a.b, false, bb.b,
                                                  (short)0, acc, false, false);
  }
#pragma unroll
  for (int r = 0; r < 8; ++r) {
    int row = ti * 16 + hl * 8 + r;
    ctx[(size_t)(b * SS + row) * HH + h * DH + d] = f2bf(acc[r]);
  }
}

// ---------------- embeddings + LayerNorm (one block / token) ----------------
__global__ __launch_bounds__(256) void embed_ln_k(
    const int* __restrict__ ids, const float* __restrict__ we,
    const float* __restrict__ pe, const float* __restrict__ te,
    const float* __restrict__ g, const float* __restrict__ bta,
    float* __restrict__ x, unsigned short* __restrict__ xb) {
  const int tok = blockIdx.x;
  const int s = tok % SS;
  const int id = ids[tok];
  __shared__ float red[256];
  float e[3];
  float sum = 0.0f;
#pragma unroll
  for (int t = 0; t < 3; ++t) {
    int i = threadIdx.x + t * 256;
    e[t] = we[(size_t)id * HH + i] + pe[(size_t)s * HH + i] + te[i];
    sum += e[t];
  }
  red[threadIdx.x] = sum;
  __syncthreads();
  for (int o = 128; o > 0; o >>= 1) {
    if (threadIdx.x < o) red[threadIdx.x] += red[threadIdx.x + o];
    __syncthreads();
  }
  float mu = red[0] / HH;
  __syncthreads();
  float vs = 0.0f;
#pragma unroll
  for (int t = 0; t < 3; ++t) { float dd = e[t] - mu; vs += dd * dd; }
  red[threadIdx.x] = vs;
  __syncthreads();
  for (int o = 128; o > 0; o >>= 1) {
    if (threadIdx.x < o) red[threadIdx.x] += red[threadIdx.x + o];
    __syncthreads();
  }
  float inv = rsqrtf(red[0] / HH + 1e-12f);
#pragma unroll
  for (int t = 0; t < 3; ++t) {
    int i = threadIdx.x + t * 256;
    float y = (e[t] - mu) * inv * g[i] + bta[i];
    x[(size_t)tok * HH + i] = y;
    xb[(size_t)tok * HH + i] = f2bf(y);
  }
}

// ---------------- residual add + LayerNorm: x = LN(x + t0); xb = bf16(x) ----------------
__global__ __launch_bounds__(256) void add_ln_k(
    const float* __restrict__ t0, const float* __restrict__ g,
    const float* __restrict__ bta, float* __restrict__ x,
    unsigned short* __restrict__ xb) {
  const int tok = blockIdx.x;
  __shared__ float red[256];
  float e[3];
  float sum = 0.0f;
#pragma unroll
  for (int t = 0; t < 3; ++t) {
    int i = threadIdx.x + t * 256;
    e[t] = x[(size_t)tok * HH + i] + t0[(size_t)tok * HH + i];
    sum += e[t];
  }
  red[threadIdx.x] = sum;
  __syncthreads();
  for (int o = 128; o > 0; o >>= 1) {
    if (threadIdx.x < o) red[threadIdx.x] += red[threadIdx.x + o];
    __syncthreads();
  }
  float mu = red[0] / HH;
  __syncthreads();
  float vs = 0.0f;
#pragma unroll
  for (int t = 0; t < 3; ++t) { float dd = e[t] - mu; vs += dd * dd; }
  red[threadIdx.x] = vs;
  __syncthreads();
  for (int o = 128; o > 0; o >>= 1) {
    if (threadIdx.x < o) red[threadIdx.x] += red[threadIdx.x + o];
    __syncthreads();
  }
  float inv = rsqrtf(red[0] / HH + 1e-12f);
#pragma unroll
  for (int t = 0; t < 3; ++t) {
    int i = threadIdx.x + t * 256;
    float y = (e[t] - mu) * inv * g[i] + bta[i];
    x[(size_t)tok * HH + i] = y;
    xb[(size_t)tok * HH + i] = f2bf(y);
  }
}

// ---------------- CRF log-likelihood (1 wave / batch, lane j = state j) ----------------
__global__ __launch_bounds__(32) void crf_llh_k(
    const float* __restrict__ em, const int* __restrict__ labels,
    const int* __restrict__ amask, const float* __restrict__ st,
    const float* __restrict__ en, const float* __restrict__ tr,
    float* __restrict__ out) {
  const int b = blockIdx.x, lane = threadIdx.x;
  const float* E = em + (size_t)b * SS * LL;
  const int jj = lane < LL ? lane : 0;
  const bool act = lane < LL;
  float alpha = act ? (st[jj] + E[jj]) : -3.0e38f;
  float num = 0.0f;
  int prev = 0;
  if (lane == 0) {
    int t0g = labels[b * SS];
    if (t0g < 0) t0g = 0;
    prev = t0g;
    num = st[t0g] + E[t0g];
  }
  for (int t = 1; t < SS; ++t) {
    float m = (float)amask[b * SS + t];
    float vals[LL];
    float best = -3.0e38f;
#pragma unroll
    for (int i = 0; i < LL; ++i) {
      float ai = __shfl(alpha, i, 32);
      float v = ai + tr[i * LL + jj];
      vals[i] = v;
      best = fmaxf(best, v);
    }
    float ss = 0.0f;
#pragma unroll
    for (int i = 0; i < LL; ++i) ss += __expf(vals[i] - best);
    float nxt = best + __logf(ss) + E[(size_t)t * LL + jj];
    if (act && m > 0.0f) alpha = nxt;
    if (lane == 0) {
      int tg = labels[b * SS + t];
      if (tg < 0) tg = 0;
      num += (tr[prev * LL + tg] + E[(size_t)t * LL + tg]) * m;
      if (m > 0.0f) prev = tg;
    }
  }
  if (lane == 0) num += en[prev];
  float fin = act ? (alpha + en[jj]) : -3.0e38f;
  float mx = fin;
  for (int o = 16; o > 0; o >>= 1) mx = fmaxf(mx, __shfl_xor(mx, o, 32));
  float se = (fin > -1.0e38f) ? __expf(fin - mx) : 0.0f;
  for (int o = 16; o > 0; o >>= 1) se += __shfl_xor(se, o, 32);
  float logZ = mx + __logf(se);
  float numb = __shfl(num, 0, 32);
  if (lane == 0) out[b] = numb - logZ;
}

// ---------------- Viterbi decode (1 wave / batch) ----------------
__global__ __launch_bounds__(32) void crf_viterbi_k(
    const float* __restrict__ em, const int* __restrict__ amask,
    const float* __restrict__ st, const float* __restrict__ en,
    const float* __restrict__ tr, int* __restrict__ hist,
    float* __restrict__ predf) {
  const int b = blockIdx.x, lane = threadIdx.x;
  const float* E = em + (size_t)b * SS * LL;
  const int jj = lane < LL ? lane : 0;
  const bool act = lane < LL;
  float score = act ? (st[jj] + E[jj]) : -3.0e38f;
  int* HB = hist + (size_t)b * (SS - 1) * LL;
  for (int t = 1; t < SS; ++t) {
    float m = (float)amask[b * SS + t];
    float best = -3.0e38f;
    int bi = 0;
#pragma unroll
    for (int i = 0; i < LL; ++i) {
      float ai = __shfl(score, i, 32);
      float v = ai + tr[i * LL + jj];
      if (v > best) { best = v; bi = i; }   // first max wins (jnp.argmax)
    }
    float nxt = best + E[(size_t)t * LL + jj];
    int hv = (m > 0.0f) ? bi : jj;
    if (act) HB[(size_t)(t - 1) * LL + lane] = hv;
    if (act && m > 0.0f) score = nxt;
  }
  float fin = act ? (score + en[jj]) : -3.0e38f;
  int last = 0;
  float bb = -3.0e38f;
  for (int i = 0; i < LL; ++i) {
    float v = __shfl(fin, i, 32);
    if (lane == 0 && v > bb) { bb = v; last = i; }
  }
  if (lane == 0) {
    float* P = predf + (size_t)b * SS;
    int tag = last;
    P[SS - 1] = (float)tag;
    for (int t = SS - 2; t >= 0; --t) {
      tag = HB[(size_t)t * LL + tag];
      P[t] = (float)tag;
    }
  }
}

__global__ void finalize_loss_k(const float* __restrict__ llh, float* __restrict__ out) {
  if (threadIdx.x == 0 && blockIdx.x == 0) {
    float s = 0.0f;
    for (int b = 0; b < BB; ++b) s += llh[b];
    out[0] = -s / (float)BB;
  }
}

// =====================================================================
extern "C" void kernel_launch(void* const* d_in, const int* in_sizes, int n_in,
                              void* d_out, int out_size, void* d_ws, size_t ws_size,
                              hipStream_t stream) {
  (void)in_sizes; (void)n_in; (void)out_size; (void)ws_size;

  // ---- inputs (top-level in setup_inputs order; params flattened jax-tree style:
  //      dict keys sorted alphabetically, lists in order) ----
  const int* input_ids = (const int*)d_in[0];
  const int* attn_mask = (const int*)d_in[1];
  const int* labels    = (const int*)d_in[2];
  const float* cls_b     = (const float*)d_in[3];
  const float* cls_w     = (const float*)d_in[4];
  const float* crf_end   = (const float*)d_in[5];
  const float* crf_start = (const float*)d_in[6];
  const float* crf_trans = (const float*)d_in[7];
  const float* emb_ln_b  = (const float*)d_in[8];
  const float* emb_ln_g  = (const float*)d_in[9];
  const int LBASE = 10;  // per layer (sorted): f1_b f1_w f2_b f2_w k_b k_w ln1_b ln1_g
                         //                     ln2_b ln2_g o_b o_w q_b q_w v_b v_w
  const float* pos_emb  = (const float*)d_in[LBASE + NLAYER * 16 + 0];
  const float* type_emb = (const float*)d_in[LBASE + NLAYER * 16 + 1];
  const float* word_emb = (const float*)d_in[LBASE + NLAYER * 16 + 2];

  // ---- workspace carve (~140 MB total) ----
  char* wp = (char*)d_ws;
  auto carve = [&](size_t bytes) -> void* {
    void* p = (void*)wp;
    wp += (bytes + 255) & ~(size_t)255;
    return p;
  };
  float*          x    = (float*)carve((size_t)MTOK * HH * 4);            // fp32 activations
  unsigned short* xb   = (unsigned short*)carve((size_t)MTOK * HH * 2);   // bf16 shadow
  unsigned short* qb   = (unsigned short*)carve((size_t)MTOK * HH * 2);
  unsigned short* kb   = (unsigned short*)carve((size_t)MTOK * HH * 2);
  unsigned short* vb   = (unsigned short*)carve((size_t)MTOK * HH * 2);
  unsigned short* attn = (unsigned short*)carve((size_t)BB * NHEAD * SS * SS * 2);
  unsigned short* ctxb = (unsigned short*)carve((size_t)MTOK * HH * 2);
  float*          t0   = (float*)carve((size_t)MTOK * HH * 4);
  unsigned short* hb   = (unsigned short*)carve((size_t)MTOK * FFN * 2);
  unsigned short* wst  = (unsigned short*)carve((size_t)HH * FFN * 2);    // weight staging
  float*          emis = (float*)carve((size_t)MTOK * LL * 4);
  float*          llh  = (float*)carve((size_t)BB * 4);
  int*            hist = (int*)carve((size_t)BB * (SS - 1) * LL * 4);

  auto cast = [&](const float* src, unsigned short* dst, size_t n) {
    int blocks = (int)((n + 255) / 256);
    if (blocks > 4096) blocks = 4096;
    cast_f32_to_bf16_k<<<blocks, 256, 0, stream>>>(src, dst, n);
  };
  auto gemm_grid = [](int N) { return dim3((unsigned)((N + 63) / 64), MTOK / 64); };

  // ---- embeddings + LN ----
  embed_ln_k<<<MTOK, 256, 0, stream>>>(input_ids, word_emb, pos_emb, type_emb,
                                       emb_ln_g, emb_ln_b, x, xb);

  // ---- transformer layers ----
  for (int l = 0; l < NLAYER; ++l) {
    const float* f1_b  = (const float*)d_in[LBASE + l * 16 + 0];
    const float* f1_w  = (const float*)d_in[LBASE + l * 16 + 1];
    const float* f2_b  = (const float*)d_in[LBASE + l * 16 + 2];
    const float* f2_w  = (const float*)d_in[LBASE + l * 16 + 3];
    const float* k_b   = (const float*)d_in[LBASE + l * 16 + 4];
    const float* k_w   = (const float*)d_in[LBASE + l * 16 + 5];
    const float* ln1_b = (const float*)d_in[LBASE + l * 16 + 6];
    const float* ln1_g = (const float*)d_in[LBASE + l * 16 + 7];
    const float* ln2_b = (const float*)d_in[LBASE + l * 16 + 8];
    const float* ln2_g = (const float*)d_in[LBASE + l * 16 + 9];
    const float* o_b   = (const float*)d_in[LBASE + l * 16 + 10];
    const float* o_w   = (const float*)d_in[LBASE + l * 16 + 11];
    const float* q_b   = (const float*)d_in[LBASE + l * 16 + 12];
    const float* q_w   = (const float*)d_in[LBASE + l * 16 + 13];
    const float* v_b   = (const float*)d_in[LBASE + l * 16 + 14];
    const float* v_w   = (const float*)d_in[LBASE + l * 16 + 15];

    // Q, K, V projections (bf16 outputs)
    cast(q_w, wst, (size_t)HH * HH);
    gemm_bf16_wmma<1><<<gemm_grid(HH), 128, 0, stream>>>(xb, wst, q_b, qb, MTOK, HH, HH);
    cast(k_w, wst, (size_t)HH * HH);
    gemm_bf16_wmma<1><<<gemm_grid(HH), 128, 0, stream>>>(xb, wst, k_b, kb, MTOK, HH, HH);
    cast(v_w, wst, (size_t)HH * HH);
    gemm_bf16_wmma<1><<<gemm_grid(HH), 128, 0, stream>>>(xb, wst, v_b, vb, MTOK, HH, HH);

    // attention
    attn_scores_k<<<dim3(SS / 16, SS / 16, BB * NHEAD), 32, 0, stream>>>(qb, kb, attn_mask, attn);
    softmax_rows_k<<<BB * NHEAD * SS, 32, 0, stream>>>(attn);
    attn_ctx_k<<<dim3(SS / 16, DH / 16, BB * NHEAD), 32, 0, stream>>>(attn, vb, ctxb);

    // output projection + residual LN
    cast(o_w, wst, (size_t)HH * HH);
    gemm_bf16_wmma<0><<<gemm_grid(HH), 128, 0, stream>>>(ctxb, wst, o_b, t0, MTOK, HH, HH);
    add_ln_k<<<MTOK, 256, 0, stream>>>(t0, ln1_g, ln1_b, x, xb);

    // FFN: gelu fused in GEMM1 epilogue
    cast(f1_w, wst, (size_t)HH * FFN);
    gemm_bf16_wmma<2><<<gemm_grid(FFN), 128, 0, stream>>>(xb, wst, f1_b, hb, MTOK, FFN, HH);
    cast(f2_w, wst, (size_t)FFN * HH);
    gemm_bf16_wmma<0><<<gemm_grid(HH), 128, 0, stream>>>(hb, wst, f2_b, t0, MTOK, HH, FFN);
    add_ln_k<<<MTOK, 256, 0, stream>>>(t0, ln2_g, ln2_b, x, xb);
  }

  // ---- classifier emissions [MTOK, L] ----
  cast(cls_w, wst, (size_t)HH * LL);
  gemm_bf16_wmma<0><<<gemm_grid(LL), 128, 0, stream>>>(xb, wst, cls_b, emis, MTOK, LL, HH);

  // ---- CRF loss + Viterbi decode ----
  float* outf = (float*)d_out;  // [0] = loss, [1..] = preds as float
  crf_llh_k<<<BB, 32, 0, stream>>>(emis, labels, attn_mask, crf_start, crf_end, crf_trans, llh);
  crf_viterbi_k<<<BB, 32, 0, stream>>>(emis, attn_mask, crf_start, crf_end, crf_trans,
                                       hist, outf + 1);
  finalize_loss_k<<<1, 32, 0, stream>>>(llh, outf);
}
